// CTR_RNN_67680094650639
// MI455X (gfx1250) — matrix operand order repaired and checked by reading, not compiled, not run on hardware
//
#include <hip/hip_runtime.h>

// ---------------- problem constants ----------------
#define Bn   256
#define Fn   32
#define Ln   50
#define Hn   128
#define G3   384          // 3*H
#define NF   8192         // B*F
#define TM   16           // batch rows per workgroup
#define NCT  24           // 384/16 column tiles
#define NKC  4            // 128/32 k-chunks
#define XPAD 8            // halfs of row padding (keeps 16B align, spreads banks)
#define GPAD 12           // floats of row padding for gate buffers

typedef __attribute__((ext_vector_type(16))) _Float16 v16h;
typedef __attribute__((ext_vector_type(8)))  _Float16 v8h;
typedef __attribute__((ext_vector_type(8)))  float    v8f;

__device__ __forceinline__ v16h cat8(v8h lo, v8h hi) {
  return __builtin_shufflevector(lo, hi, 0,1,2,3,4,5,6,7,8,9,10,11,12,13,14,15);
}
__device__ __forceinline__ v8f wmma_f16(v16h a, v16h b, v8f c) {
  // D = A(16x32 f16) * B(32x16 f16) + C(16x16 f32)
  return __builtin_amdgcn_wmma_f32_16x16x32_f16(false, a, false, b, (short)0, c,
                                                false, false);
}
// A-fragment: lane m=lane&15 owns row m; elements 0..15 -> K = kc*32+hi*16..+15
// (contiguous, matching the B-fragment packing)
__device__ __forceinline__ v16h load_a_frag(const _Float16* rowptr, int kc, int hi) {
  const v8h* p1 = (const v8h*)(rowptr + kc * 32 + hi * 16);
  const v8h* p2 = (const v8h*)(rowptr + kc * 32 + hi * 16 + 8);
  return cat8(*p1, *p2);
}
// B-fragment pre-packed per-lane contiguous (16 halfs)
__device__ __forceinline__ v16h load_b_frag(const _Float16* p) {
  const v8h* q = (const v8h*)p;
  return cat8(q[0], q[1]);
}

__device__ __forceinline__ float sigmoidf_fast(float x) {
  return 1.f / (1.f + __expf(-x));
}
__device__ __forceinline__ float tanhf_fast(float x) {
#if __has_builtin(__builtin_amdgcn_tanhf)
  return __builtin_amdgcn_tanhf(x);
#else
  return 1.f - 2.f / (__expf(2.f * x) + 1.f);   // stable at both tails
#endif
}

// =====================================================================
// Fused GRU: embed-gather + gi GEMM + recurrent GEMM + gates, 50 steps,
// all weights LDS-resident in WMMA B-fragment layout. 16 rows / WG, 8 waves.
// Software-pipelined: embed row for t+1 prefetched into VGPRs during the
// WMMA + gate work of step t. Captures h at t == len-1 into out_last.
// =====================================================================
__global__ __launch_bounds__(256) void gru_fused_kernel(
    const int*   __restrict__ src,     // [N, L]
    const float* __restrict__ embed,   // [V, H]
    const float* __restrict__ Wih,     // [3H, H] row-major
    const float* __restrict__ Whh,     // [3H, H]
    const float* __restrict__ bih,     // [3H]
    const float* __restrict__ bhh,     // [3H]
    float*       __restrict__ out_last,// [N, H]
    int N)
{
  __shared__ __align__(16) _Float16 wih_f[NCT * NKC * 32 * 16]; // 96 KB
  __shared__ __align__(16) _Float16 whh_f[NCT * NKC * 32 * 16]; // 96 KB
  __shared__ __align__(16) _Float16 x16[TM][Hn + XPAD];
  __shared__ __align__(16) _Float16 h16[TM][Hn + XPAD];
  __shared__ float h32[TM][Hn];
  __shared__ float gi32[TM][G3 + GPAD];
  __shared__ float gh32[TM][G3 + GPAD];
  __shared__ float bihs[G3];
  __shared__ float bhhs[G3];
  __shared__ int   toks_all[TM][Ln];
  __shared__ int   lens[TM];

  const int tid  = threadIdx.x;
  const int lane = tid & 31;
  const int wave = tid >> 5;
  const int row0 = blockIdx.x * TM;

  // ---- pack Wih/Whh into B-fragment order: B[k, g] = W[g, k] ----
  for (int idx = tid; idx < NCT * NKC * 32; idx += 256) {
    const int l  = idx & 31;
    const int kc = (idx >> 5) & 3;
    const int ct = idx >> 7;
    const int g  = ct * 16 + (l & 15);
    const int k0 = kc * 32 + (l >> 4) * 16;
    const float* wi = Wih + g * Hn + k0;
    const float* wh = Whh + g * Hn + k0;
    _Float16* di = &wih_f[idx * 16];
    _Float16* dh = &whh_f[idx * 16];
#pragma unroll
    for (int e = 0; e < 16; ++e) { di[e] = (_Float16)wi[e]; dh[e] = (_Float16)wh[e]; }
  }
  for (int i = tid; i < G3; i += 256) { bihs[i] = bih[i]; bhhs[i] = bhh[i]; }
  // all tokens for this workgroup's rows (L2-resident; removes per-step loads)
  for (int i = tid; i < TM * Ln; i += 256)
    toks_all[i / Ln][i % Ln] = src[row0 * Ln + i];
  if (tid < TM) {
    int cnt = 0;
    const int* s = src + (row0 + tid) * Ln;
    for (int l = 0; l < Ln; ++l) cnt += (s[l] != 0);
    lens[tid] = cnt;
  }
  for (int i = tid; i < TM * Hn; i += 256) {
    const int r = i >> 7, c = i & (Hn - 1);
    h32[r][c] = 0.f;
    h16[r][c] = (_Float16)0.f;
  }
  __syncthreads();

  const int r  = tid >> 4;           // gather/gate row (0..15)
  const int c0 = (tid & 15) * 8;     // gather/gate col group
  const int m  = lane & 15;          // WMMA A row
  const int hi = lane >> 4;

  // prefetch embedding row chunk for t=0
  float4 pe0, pe1;
  {
    const float4* ev = (const float4*)(embed + (size_t)toks_all[r][0] * Hn + c0);
    pe0 = ev[0]; pe1 = ev[1];
  }

  for (int t = 0; t < Ln; ++t) {
    // commit prefetched x_t (f32 -> f16) into LDS
    x16[r][c0 + 0] = (_Float16)pe0.x; x16[r][c0 + 1] = (_Float16)pe0.y;
    x16[r][c0 + 2] = (_Float16)pe0.z; x16[r][c0 + 3] = (_Float16)pe0.w;
    x16[r][c0 + 4] = (_Float16)pe1.x; x16[r][c0 + 5] = (_Float16)pe1.y;
    x16[r][c0 + 6] = (_Float16)pe1.z; x16[r][c0 + 7] = (_Float16)pe1.w;
    __syncthreads();   // x16 + h16(t-1 gates) visible to all waves

    // prefetch t+1 embedding (overlaps the GEMM + gates below)
    if (t + 1 < Ln) {
      const float4* ev =
          (const float4*)(embed + (size_t)toks_all[r][t + 1] * Hn + c0);
      pe0 = ev[0]; pe1 = ev[1];
    }

    // ---- dual GEMM: gi = x @ Wih^T, gh = h @ Whh^T (3 tiles per wave) ----
#pragma unroll
    for (int q = 0; q < 3; ++q) {
      const int ct = wave + q * 8;
      v8f acc_i = {};
      v8f acc_h = {};
#pragma unroll
      for (int kc = 0; kc < NKC; ++kc) {
        const v16h ax = load_a_frag(&x16[m][0], kc, hi);
        const v16h ah = load_a_frag(&h16[m][0], kc, hi);
        const v16h bi = load_b_frag(&wih_f[((ct * NKC + kc) * 32 + lane) * 16]);
        const v16h bh = load_b_frag(&whh_f[((ct * NKC + kc) * 32 + lane) * 16]);
        acc_i = wmma_f16(ax, bi, acc_i);
        acc_h = wmma_f16(ah, bh, acc_h);
      }
      const int n = lane & 15;
#pragma unroll
      for (int v = 0; v < 8; ++v) {
        gi32[v + hi * 8][ct * 16 + n] = acc_i[v];
        gh32[v + hi * 8][ct * 16 + n] = acc_h[v];
      }
    }
    __syncthreads();   // gi32/gh32 complete

    { // ---- gates + state update (torch order r,z,n) ----
      const bool last = (t == lens[r] - 1);
#pragma unroll
      for (int i = 0; i < 8; ++i) {
        const int c = c0 + i;
        const float gr  = gi32[r][c]          + bihs[c]          + gh32[r][c]          + bhhs[c];
        const float gz  = gi32[r][Hn + c]     + bihs[Hn + c]     + gh32[r][Hn + c]     + bhhs[Hn + c];
        const float gin = gi32[r][2 * Hn + c] + bihs[2 * Hn + c];
        const float ghn = gh32[r][2 * Hn + c] + bhhs[2 * Hn + c];
        const float rr = sigmoidf_fast(gr);
        const float zz = sigmoidf_fast(gz);
        const float nn = tanhf_fast(gin + rr * ghn);
        const float hp = h32[r][c];
        const float hnew = (1.f - zz) * nn + zz * hp;
        h32[r][c] = hnew;
        h16[r][c] = (_Float16)hnew;
        if (last) out_last[(size_t)(row0 + r) * Hn + c] = hnew;
      }
    }
    // no barrier here: next iteration's x16 commit + barrier covers h16 WAR/RAW
  }
  if (tid < TM && lens[tid] == 0) {   // all-pad row: reference yields zeros
    for (int c = 0; c < Hn; ++c) out_last[(size_t)(row0 + tid) * Hn + c] = 0.f;
  }
}

// =====================================================================
// sf = [self_rep | friend_last] @ W_friend^T   (sf stays in LDS)
// v  = sf @ W_beta                              -> vbeta [NF, H]
// 16 rows / WG, 8 waves = 8 column tiles of 16.
// =====================================================================
__global__ __launch_bounds__(256) void sfv_kernel(
    const float* __restrict__ self_last,   // [B, H]
    const float* __restrict__ friend_last, // [NF, H]
    const float* __restrict__ W_friend,    // [H, 2H]
    const float* __restrict__ W_beta,      // [H, H]
    float*       __restrict__ vbeta)       // [NF, H]
{
  __shared__ __align__(16) _Float16 wf_f[8 * 8 * 32 * 16]; // 64 KB
  __shared__ __align__(16) _Float16 wb_f[8 * 4 * 32 * 16]; // 32 KB
  __shared__ __align__(16) _Float16 acat[TM][256 + XPAD];
  __shared__ __align__(16) _Float16 sf16[TM][Hn + XPAD];

  const int tid  = threadIdx.x;
  const int lane = tid & 31;
  const int wave = tid >> 5;
  const int row0 = blockIdx.x * TM;

  // W_friend fragments: B[k, j] = W_friend[j, k], K=256 (8 kc), N=128 (8 ct)
  for (int idx = tid; idx < 8 * 8 * 32; idx += 256) {
    const int l  = idx & 31;
    const int kc = (idx >> 5) & 7;
    const int ct = idx >> 8;
    const int j  = ct * 16 + (l & 15);
    const int k0 = kc * 32 + (l >> 4) * 16;
    const float* s = W_friend + j * 256 + k0;
    _Float16* d = &wf_f[idx * 16];
#pragma unroll
    for (int e = 0; e < 16; ++e) d[e] = (_Float16)s[e];
  }
  // W_beta fragments: B[k, h] = W_beta[k, h], K=128 (4 kc), N=128 (8 ct)
  for (int idx = tid; idx < 8 * 4 * 32; idx += 256) {
    const int l  = idx & 31;
    const int kc = (idx >> 5) & 3;
    const int ct = idx >> 7;
    const int h  = ct * 16 + (l & 15);
    const int k0 = kc * 32 + (l >> 4) * 16;
    _Float16* d = &wb_f[idx * 16];
#pragma unroll
    for (int e = 0; e < 16; ++e) d[e] = (_Float16)W_beta[(k0 + e) * Hn + h];
  }
  // A = concat(self_rep, friend_last) rows
  for (int i = tid; i < TM * 256; i += 256) {
    const int r = i >> 8, c = i & 255;
    const int n = row0 + r;
    const float v = (c < Hn) ? self_last[(n >> 5) * Hn + c]
                             : friend_last[(size_t)n * Hn + (c - Hn)];
    acat[r][c] = (_Float16)v;
  }
  __syncthreads();

  const int m  = lane & 15;
  const int hi = lane >> 4;
  { // GEMM 1: sf
    const int ct = wave;
    v8f acc = {};
#pragma unroll
    for (int kc = 0; kc < 8; ++kc) {
      const v16h a = load_a_frag(&acat[m][0], kc, hi);
      const v16h b = load_b_frag(&wf_f[((ct * 8 + kc) * 32 + lane) * 16]);
      acc = wmma_f16(a, b, acc);
    }
    const int n = lane & 15;
#pragma unroll
    for (int v = 0; v < 8; ++v) sf16[v + hi * 8][ct * 16 + n] = (_Float16)acc[v];
  }
  __syncthreads();
  { // GEMM 2: v = sf @ W_beta
    const int ct = wave;
    v8f acc = {};
#pragma unroll
    for (int kc = 0; kc < 4; ++kc) {
      const v16h a = load_a_frag(&sf16[m][0], kc, hi);
      const v16h b = load_b_frag(&wb_f[((ct * 4 + kc) * 32 + lane) * 16]);
      acc = wmma_f16(a, b, acc);
    }
    const int n = lane & 15;
#pragma unroll
    for (int v = 0; v < 8; ++v)
      vbeta[(size_t)(row0 + v + hi * 8) * Hn + ct * 16 + n] = acc[v];
  }
}

// =====================================================================
// friendship[n] = sum_l softplus(embed[tok]·v_n) * exp(1 - t/TAU) * (tok!=0)
// One wave per n row; 32-lane dot of H=128 via shfl_xor reduction.
// =====================================================================
__global__ __launch_bounds__(256) void score_kernel(
    const int*   __restrict__ common_src,   // [NF, L]
    const float* __restrict__ common_time,  // [NF, L]
    const float* __restrict__ embed,        // [V, H]
    const float* __restrict__ vbeta,        // [NF, H]
    float*       __restrict__ friendship)   // [NF]
{
  const int lane = threadIdx.x & 31;
  const int wave = threadIdx.x >> 5;
  const int n = blockIdx.x * 8 + wave;

  const float4 v = *(const float4*)(vbeta + (size_t)n * Hn + lane * 4);
  const int*   src = common_src  + n * Ln;
  const float* tmv = common_time + n * Ln;

  float acc = 0.f;
  for (int l = 0; l < Ln; ++l) {
    const int tok = src[l];
    if (tok != 0) {
      const float4 e = *(const float4*)(embed + (size_t)tok * Hn + lane * 4);
      float p = e.x * v.x + e.y * v.y + e.z * v.z + e.w * v.w;
#pragma unroll
      for (int k = 16; k >= 1; k >>= 1) p += __shfl_xor(p, k, 32);
      const float sp = (p > 20.f) ? p : __logf(1.f + __expf(p));
      const float w  = __expf(1.f - tmv[l] * (1.0f / 1.0e6f));
      acc += sp * w;
    }
  }
  if (lane == 0) friendship[n] = acc;
}

// =====================================================================
// masked softmax over F, attention-aggregate friends, final linear + ReLU
// one workgroup (128 threads) per batch element.
// =====================================================================
__global__ __launch_bounds__(128) void attn_out_kernel(
    const float* __restrict__ friendship,   // [B*F]
    const int*   __restrict__ friend_num,   // [B]
    const float* __restrict__ friend_last,  // [NF, H]
    const float* __restrict__ self_last,    // [B, H]
    const float* __restrict__ W_out,        // [H, 2H]
    const float* __restrict__ b_out,        // [H]
    float*       __restrict__ out)          // [B, H]
{
  __shared__ float attn_s[Fn];
  __shared__ float agg_s[Hn];
  __shared__ float self_s[Hn];

  const int b   = blockIdx.x;
  const int tid = threadIdx.x;
  const int nf  = friend_num[b];

  if (tid < Fn) { // wave 0: masked softmax over the 32 slots
    const float valid = (tid < nf) ? 1.f : 0.f;
    const float fr = friendship[b * Fn + tid] * valid;  // zeros still enter softmax
    float mx = fr;
#pragma unroll
    for (int k = 16; k >= 1; k >>= 1) mx = fmaxf(mx, __shfl_xor(mx, k, 32));
    const float e = __expf(fr - mx);
    float s = e;
#pragma unroll
    for (int k = 16; k >= 1; k >>= 1) s += __shfl_xor(s, k, 32);
    attn_s[tid] = e / s;
  }
  self_s[tid] = self_last[b * Hn + tid];
  __syncthreads();

  float agg = 0.f;
  for (int f = 0; f < Fn; ++f) {
    const float valid = (f < nf) ? 1.f : 0.f;
    agg += attn_s[f] * valid * friend_last[(size_t)(b * Fn + f) * Hn + tid];
  }
  agg_s[tid] = agg;
  __syncthreads();

  float acc = b_out[tid];
  const float* wrow = W_out + tid * (2 * Hn);
#pragma unroll 4
  for (int k = 0; k < Hn; ++k) acc += self_s[k] * wrow[k];
#pragma unroll 4
  for (int k = 0; k < Hn; ++k) acc += agg_s[k] * wrow[Hn + k];
  out[b * Hn + tid] = fmaxf(acc, 0.f);
}

// =====================================================================
extern "C" void kernel_launch(void* const* d_in, const int* in_sizes, int n_in,
                              void* d_out, int out_size, void* d_ws, size_t ws_size,
                              hipStream_t stream) {
  (void)in_sizes; (void)n_in; (void)out_size; (void)ws_size;
  const int*   self_src    = (const int*)  d_in[0];
  const int*   common_src  = (const int*)  d_in[1];
  const float* common_time = (const float*)d_in[2];
  const int*   friend_src  = (const int*)  d_in[3];
  const int*   friend_num  = (const int*)  d_in[4];
  const float* embed       = (const float*)d_in[5];
  const float* Wih_s = (const float*)d_in[6];
  const float* Whh_s = (const float*)d_in[7];
  const float* bih_s = (const float*)d_in[8];
  const float* bhh_s = (const float*)d_in[9];
  const float* Wih_f = (const float*)d_in[10];
  const float* Whh_f = (const float*)d_in[11];
  const float* bih_f = (const float*)d_in[12];
  const float* bhh_f = (const float*)d_in[13];
  const float* W_friend = (const float*)d_in[14];
  const float* W_beta   = (const float*)d_in[15];
  const float* W_out    = (const float*)d_in[16];
  const float* b_out    = (const float*)d_in[17];

  float* ws          = (float*)d_ws;
  float* self_last   = ws;                            // 256*128
  float* friend_last = self_last + Bn * Hn;           // 8192*128
  float* vbeta       = friend_last + NF * Hn;         // 8192*128
  float* friendship  = vbeta + NF * Hn;               // 8192

  gru_fused_kernel<<<Bn / TM, 256, 0, stream>>>(
      self_src, embed, Wih_s, Whh_s, bih_s, bhh_s, self_last, Bn);
  gru_fused_kernel<<<NF / TM, 256, 0, stream>>>(
      friend_src, embed, Wih_f, Whh_f, bih_f, bhh_f, friend_last, NF);
  sfv_kernel<<<NF / TM, 256, 0, stream>>>(
      self_last, friend_last, W_friend, W_beta, vbeta);
  score_kernel<<<NF / 8, 256, 0, stream>>>(
      common_src, common_time, embed, vbeta, friendship);
  attn_out_kernel<<<Bn, 128, 0, stream>>>(
      friendship, friend_num, friend_last, self_last, W_out, b_out, (float*)d_out);
}
